// NNRankModel_35828617183464
// MI455X (gfx1250) — matrix operand order repaired and checked by reading, not compiled, not run on hardware
//
#include <hip/hip_runtime.h>
#include <hip/hip_bf16.h>
#include <math.h>

typedef _Float16 v16h __attribute__((ext_vector_type(16)));
typedef _Float16 v8h  __attribute__((ext_vector_type(8)));
typedef float    v8f  __attribute__((ext_vector_type(8)));
typedef float    v4f  __attribute__((ext_vector_type(4)));

#define F_FIELDS 40
#define EMB      8
#define NPAIRS   780          // 40*39/2
#define DNN_IN   12480        // 2*780*8
#define BATCH    8192
#define BN_EPS   1e-5f

// ---------------------------------------------------------------------------
// B-operand: lane holds 16 consecutive K values.
__device__ __forceinline__ v16h load_b16(const _Float16* __restrict__ p) {
  v8h lo = *reinterpret_cast<const v8h*>(p);
  v8h hi = *reinterpret_cast<const v8h*>(p + 8);
  v16h r;
#pragma unroll
  for (int i = 0; i < 8; ++i) { r[i] = lo[i]; r[i + 8] = hi[i]; }
  return r;
}
// A-operand: lane holds 16 consecutive K values.
__device__ __forceinline__ v16h load_a16(const _Float16* __restrict__ p) {
  v8h lo = *reinterpret_cast<const v8h*>(p);
  v8h hi = *reinterpret_cast<const v8h*>(p + 8);
  v16h r;
#pragma unroll
  for (int i = 0; i < 8; ++i) { r[i] = lo[i]; r[i + 8] = hi[i]; }
  return r;
}

// ---------------------------------------------------------------------------
// Kernel 1: BN + SENet + bilinear-all feature construction.
// One block (256 thr) per batch row. Produces h[b][12480] in f16:
//   h[0:6240]     = A[i]*A[j] * (V0[i]@W^T)*V0[j]   (SENet branch)
//   h[6240:12480] =             (V0[i]@W^T)*V0[j]   (raw branch)
__global__ __launch_bounds__(256) void fibinet_features(
    const float* __restrict__ emb, const float* __restrict__ bnw,
    const float* __restrict__ bnb, const float* __restrict__ bnm,
    const float* __restrict__ bnv, const float* __restrict__ sew1,
    const float* __restrict__ sew2, const float* __restrict__ wbl,
    _Float16* __restrict__ h)
{
  __shared__ float V0[F_FIELDS * EMB];
  __shared__ float T0[F_FIELDS * EMB];
  __shared__ float Zs[F_FIELDS];
  __shared__ float ZR[8];
  __shared__ float Aex[F_FIELDS];
  __shared__ float Ws[EMB * EMB];
  __shared__ unsigned char PI[NPAIRS];
  __shared__ unsigned char PJ[NPAIRS];

  const int tid = threadIdx.x;
  const int b   = blockIdx.x;

  if (tid < F_FIELDS) {                 // pair index table
    int base = tid * 39 - tid * (tid - 1) / 2;
    for (int j = tid + 1; j < F_FIELDS; ++j) {
      PI[base + j - tid - 1] = (unsigned char)tid;
      PJ[base + j - tid - 1] = (unsigned char)j;
    }
  }
  if (tid < EMB * EMB) Ws[tid] = wbl[tid];
  if (tid < F_FIELDS * EMB) {           // BatchNorm with running stats
    float e  = emb[(size_t)b * (F_FIELDS * EMB) + tid];
    float sc = bnw[tid] * rsqrtf(bnv[tid] + BN_EPS);
    V0[tid]  = (e - bnm[tid]) * sc + bnb[tid];
  }
  __syncthreads();

  if (tid < F_FIELDS) {                 // squeeze: mean over E
    float s = 0.f;
#pragma unroll
    for (int e = 0; e < EMB; ++e) s += V0[tid * EMB + e];
    Zs[tid] = s * (1.0f / EMB);
  }
  __syncthreads();

  if (tid < 5) {                        // excitation 40 -> 5, relu
    float s = 0.f;
    for (int f = 0; f < F_FIELDS; ++f) s += Zs[f] * sew1[tid * F_FIELDS + f];
    ZR[tid] = fmaxf(s, 0.f);
  }
  __syncthreads();

  if (tid < F_FIELDS) {                 // excitation 5 -> 40, relu
    float s = 0.f;
#pragma unroll
    for (int r = 0; r < 5; ++r) s += ZR[r] * sew2[tid * 5 + r];
    Aex[tid] = fmaxf(s, 0.f);
  }
  if (tid < F_FIELDS * EMB) {           // t0[f][d] = sum_e V0[f][e]*W[d][e]
    int f = tid >> 3, d = tid & 7;
    float s = 0.f;
#pragma unroll
    for (int e = 0; e < EMB; ++e) s += V0[f * EMB + e] * Ws[d * EMB + e];
    T0[tid] = s;
  }
  __syncthreads();

  _Float16* hrow = h + (size_t)b * DNN_IN;
  for (int idx = tid; idx < NPAIRS * EMB; idx += 256) {
    int p = idx >> 3, e = idx & 7;
    int i = PI[p], j = PJ[p];
    float v = T0[i * EMB + e] * V0[j * EMB + e];
    hrow[NPAIRS * EMB + idx] = (_Float16)v;                     // raw branch
    hrow[idx]                = (_Float16)(Aex[i] * Aex[j] * v); // SE branch
  }
}

// ---------------------------------------------------------------------------
// Kernel 2: vectorized f32 -> f16 weight conversion (8 elems/thread).
__global__ void f32_to_f16_v8(const float* __restrict__ src,
                              _Float16* __restrict__ dst, int n8)
{
  int i = blockIdx.x * blockDim.x + threadIdx.x;
  if (i >= n8) return;
  v4f a = *reinterpret_cast<const v4f*>(src + (size_t)i * 8);
  v4f b = *reinterpret_cast<const v4f*>(src + (size_t)i * 8 + 4);
  v8h o;
#pragma unroll
  for (int t = 0; t < 4; ++t) { o[t] = (_Float16)a[t]; o[t + 4] = (_Float16)b[t]; }
  *reinterpret_cast<v8h*>(dst + (size_t)i * 8) = o;
}

// ---------------------------------------------------------------------------
// Kernel 3/4: WMMA GEMM   C[M][N] = act(A[M][K] * Bw[N][K]^T + bias)
// One wave computes a 32x64 tile of C: 2 A operands x 4 B operands
// -> 8 v_wmma per 32-K step, 21.3 FLOP per operand byte.
template <bool RELU>
__global__ __launch_bounds__(256) void gemm_wmma_f16(
    const _Float16* __restrict__ A,    // [M][K] row-major
    const _Float16* __restrict__ Bw,   // [N][K] row-major (out x in)
    const float*    __restrict__ bias, // [N]
    _Float16*       __restrict__ C,    // [M][N] row-major
    int M, int N, int K)
{
  const int lane    = threadIdx.x & 31;
  const int wave    = blockIdx.x * (blockDim.x >> 5) + (threadIdx.x >> 5);
  const int nGroups = N >> 6;               // 64 output cols per wave
  const int mTile   = wave / nGroups;       // 32 output rows per wave
  const int nGroup  = wave % nGroups;
  if (mTile * 32 >= M) return;

  const int row = lane & 15;                // M row (A) / N col (B) in tile
  const int hs  = lane >> 4;                // lane-half selector

  const _Float16* aBase0 = A + (size_t)(mTile * 32 +  0 + row) * K + hs * 16;
  const _Float16* aBase1 = A + (size_t)(mTile * 32 + 16 + row) * K + hs * 16;
  const _Float16* bBase0 = Bw + (size_t)(nGroup * 64 +  0 + row) * K + hs * 16;
  const _Float16* bBase1 = Bw + (size_t)(nGroup * 64 + 16 + row) * K + hs * 16;
  const _Float16* bBase2 = Bw + (size_t)(nGroup * 64 + 32 + row) * K + hs * 16;
  const _Float16* bBase3 = Bw + (size_t)(nGroup * 64 + 48 + row) * K + hs * 16;

  v8f acc00, acc01, acc02, acc03, acc10, acc11, acc12, acc13;
#pragma unroll
  for (int r = 0; r < 8; ++r) {
    acc00[r] = 0.f; acc01[r] = 0.f; acc02[r] = 0.f; acc03[r] = 0.f;
    acc10[r] = 0.f; acc11[r] = 0.f; acc12[r] = 0.f; acc13[r] = 0.f;
  }

  for (int k0 = 0; k0 < K; k0 += 32) {
    v16h a0 = load_a16(aBase0 + k0);
    v16h a1 = load_a16(aBase1 + k0);
    v16h b0 = load_b16(bBase0 + k0);
    v16h b1 = load_b16(bBase1 + k0);
    v16h b2 = load_b16(bBase2 + k0);
    v16h b3 = load_b16(bBase3 + k0);
    acc00 = __builtin_amdgcn_wmma_f32_16x16x32_f16(false, a0, false, b0, (short)0, acc00, false, false);
    acc01 = __builtin_amdgcn_wmma_f32_16x16x32_f16(false, a0, false, b1, (short)0, acc01, false, false);
    acc02 = __builtin_amdgcn_wmma_f32_16x16x32_f16(false, a0, false, b2, (short)0, acc02, false, false);
    acc03 = __builtin_amdgcn_wmma_f32_16x16x32_f16(false, a0, false, b3, (short)0, acc03, false, false);
    acc10 = __builtin_amdgcn_wmma_f32_16x16x32_f16(false, a1, false, b0, (short)0, acc10, false, false);
    acc11 = __builtin_amdgcn_wmma_f32_16x16x32_f16(false, a1, false, b1, (short)0, acc11, false, false);
    acc12 = __builtin_amdgcn_wmma_f32_16x16x32_f16(false, a1, false, b2, (short)0, acc12, false, false);
    acc13 = __builtin_amdgcn_wmma_f32_16x16x32_f16(false, a1, false, b3, (short)0, acc13, false, false);
  }

  // epilogue: bias + relu + f16 store.
  // D layout: VGPR r, lane L -> M = tileBase + r + 8*(L/16), N = colBase + s*16
  const int colBase = nGroup * 64 + (lane & 15);
  v8f* accs[8] = {&acc00, &acc01, &acc02, &acc03, &acc10, &acc11, &acc12, &acc13};
#pragma unroll
  for (int mt = 0; mt < 2; ++mt) {
#pragma unroll
    for (int s = 0; s < 4; ++s) {
      v8f* accp = accs[mt * 4 + s];
      int n = colBase + s * 16;
      float bv = bias[n];
#pragma unroll
      for (int r = 0; r < 8; ++r) {
        int m = mTile * 32 + mt * 16 + r + 8 * hs;
        float v = (*accp)[r] + bv;
        if (RELU) v = fmaxf(v, 0.f);
        C[(size_t)m * N + n] = (_Float16)v;
      }
    }
  }
}

// ---------------------------------------------------------------------------
// Kernel 5: head  out[b] = sigmoid(dot(C2[b], w3) + b3).  One wave per row.
__global__ __launch_bounds__(256) void final_head(
    const _Float16* __restrict__ C2, const float* __restrict__ w3,
    const float* __restrict__ b3, float* __restrict__ out, int M)
{
  const int lane = threadIdx.x & 31;
  const int row  = blockIdx.x * (blockDim.x >> 5) + (threadIdx.x >> 5);
  if (row >= M) return;
  const _Float16* c = C2 + (size_t)row * 512;
  float s = 0.f;
#pragma unroll
  for (int t = 0; t < 16; ++t) {
    int k = lane + 32 * t;
    s += (float)c[k] * w3[k];
  }
#pragma unroll
  for (int off = 16; off > 0; off >>= 1) s += __shfl_xor(s, off, 32);
  if (lane == 0) out[row] = 1.f / (1.f + expf(-(s + b3[0])));
}

// ---------------------------------------------------------------------------
extern "C" void kernel_launch(void* const* d_in, const int* in_sizes, int n_in,
                              void* d_out, int out_size, void* d_ws, size_t ws_size,
                              hipStream_t stream)
{
  const float* emb  = (const float*)d_in[0];
  const float* bnw  = (const float*)d_in[1];
  const float* bnb  = (const float*)d_in[2];
  const float* bnm  = (const float*)d_in[3];
  const float* bnv  = (const float*)d_in[4];
  const float* sew1 = (const float*)d_in[5];
  const float* sew2 = (const float*)d_in[6];
  const float* wbl  = (const float*)d_in[7];
  const float* w1f  = (const float*)d_in[8];
  const float* b1   = (const float*)d_in[9];
  const float* w2f  = (const float*)d_in[10];
  const float* b2   = (const float*)d_in[11];
  const float* w3   = (const float*)d_in[12];
  const float* b3   = (const float*)d_in[13];

  // workspace layout (all offsets 16B aligned)
  _Float16* h   = (_Float16*)d_ws;                      // 8192*12480
  _Float16* w1h = h   + (size_t)BATCH * DNN_IN;         // 1024*12480
  _Float16* w2h = w1h + (size_t)1024 * DNN_IN;          // 512*1024
  _Float16* c1  = w2h + (size_t)512 * 1024;             // 8192*1024
  _Float16* c2  = c1  + (size_t)BATCH * 1024;           // 8192*512

  // 1) feature construction
  fibinet_features<<<BATCH, 256, 0, stream>>>(emb, bnw, bnb, bnm, bnv,
                                              sew1, sew2, wbl, h);
  // 2) weight conversion (8 elems/thread; sizes divisible by 8)
  {
    int n1_8 = (1024 * DNN_IN) / 8;     // 1,597,440
    f32_to_f16_v8<<<(n1_8 + 255) / 256, 256, 0, stream>>>(w1f, w1h, n1_8);
    int n2_8 = (512 * 1024) / 8;        // 65,536
    f32_to_f16_v8<<<(n2_8 + 255) / 256, 256, 0, stream>>>(w2f, w2h, n2_8);
  }
  // 3) GEMM1: [8192,12480] x [12480,1024] -> relu -> c1
  //    waves = (8192/32)*(1024/64) = 4096; 8 waves/block -> 512 blocks
  gemm_wmma_f16<true><<<512, 256, 0, stream>>>(h, w1h, b1, c1,
                                               BATCH, 1024, DNN_IN);
  // 4) GEMM2: [8192,1024] x [1024,512] -> relu -> c2
  //    waves = 256*8 = 2048 -> 256 blocks
  gemm_wmma_f16<true><<<256, 256, 0, stream>>>(c1, w2h, b2, c2,
                                               BATCH, 512, 1024);
  // 5) head
  final_head<<<BATCH / 8, 256, 0, stream>>>(c2, w3, b3, (float*)d_out, BATCH);
}
